// BackwardInjectWSConv2D_63617055588702
// MI455X (gfx1250) — matrix operand order repaired and checked by reading, not compiled, not run on hardware
//
#include <hip/hip_runtime.h>
#include <hip/hip_bf16.h>

typedef __attribute__((ext_vector_type(16))) __bf16 v16bf;
typedef __attribute__((ext_vector_type(8)))  __bf16 v8bf;
typedef __attribute__((ext_vector_type(8)))  float  v8f;

#define CH   128
#define HW   56
#define IMG  (HW*HW)          // 3136
#define NB   64
#define MTOT (NB*IMG)         // 200704
#define SPLIT 49
#define PIX_PER_BLOCK 4096    // 32 rounds * 128 pixels
#define APAD 136              // bf16 row pad: 272B rows, 16B-aligned, bank stride 4
#define CPAD 132              // f32 C-tile row pad (528B)

// Desired inner-loop shape per K-substep:
//   8 DS reads, then (1 WMMA, 2 DS reads) x5, then 3 WMMAs.
// Gives every WMMA >= 2 fragment-pairs of dscnt slack instead of wait-0.
__device__ __forceinline__ void mma_pipeline_hint()
{
  __builtin_amdgcn_sched_group_barrier(0x100, 8, 0);   // 8 DS reads
  #pragma unroll
  for (int u = 0; u < 5; ++u) {
    __builtin_amdgcn_sched_group_barrier(0x008, 1, 0); // 1 WMMA
    __builtin_amdgcn_sched_group_barrier(0x100, 2, 0); // 2 DS reads
  }
  __builtin_amdgcn_sched_group_barrier(0x008, 3, 0);   // 3 WMMAs
}

// ---------------------------------------------------------------------------
// dgrad: grad_in[m, ci] = sum_{kh,kw,co} go[b, y+kh-1, x+kw-1, co] *
//                                        wstd[2-kh, 2-kw, ci, co]
// Implicit GEMM, M=200704, N=128. Per (kh,kw): stage full 128-co slab in LDS
// as bf16, then 4 K-substeps x 8 WMMAs (32 v_wmma per barrier pair).
// Writeback staged through LDS for coalesced global_store_b128.
// ---------------------------------------------------------------------------
__global__ __launch_bounds__(256) void dgrad_kernel(
    const float* __restrict__ go, const float* __restrict__ wstd,
    float* __restrict__ gin)
{
  __shared__ __align__(16) char smem[2 * 128 * APAD * sizeof(__bf16)]; // 69632B
  typedef __bf16 (*TileT)[APAD];
  TileT As = (TileT)smem;                                   // [m-row][co]
  TileT Bs = (TileT)(smem + 128 * APAD * sizeof(__bf16));   // [ci   ][co]

  const int t    = threadIdx.x;
  const int wave = t >> 5, lane = t & 31;
  const int m0   = blockIdx.x * 128;

  v8f acc[8];
  #pragma unroll
  for (int j = 0; j < 8; ++j)
    #pragma unroll
    for (int v = 0; v < 8; ++v) acc[j][v] = 0.f;

  // staging mapping: 2 threads per row, 64 contiguous floats each
  const int r    = t >> 1;
  const int half = t & 1;
  const int m  = m0 + r;
  const int b  = m / IMG;
  const int yx = m - b * IMG;
  const int y  = yx / HW;
  const int x  = yx - y * HW;

  // fragment geometry (ISA 16-bit A/B layouts)
  const int arow = wave * 16 + (lane & 15);
  const int akb  = (lane < 16) ? 0 : 8;
  const int bkb  = (lane < 16) ? 0 : 8;
  const int nn   = lane & 15;

  for (int kh = 0; kh < 3; ++kh) {
    const int sy = y + kh - 1;
    for (int kw = 0; kw < 3; ++kw) {
      const int sx = x + kw - 1;
      { // stage A: 128 rows x 128 co of shifted grad_output, f32 -> bf16
        const bool ok = (sy >= 0) && (sy < HW) && (sx >= 0) && (sx < HW);
        __bf16* dst = &As[r][half * 64];
        if (ok) {
          const float* src = go + ((size_t)((b * HW + sy) * HW + sx)) * CH + half * 64;
          #pragma unroll
          for (int i = 0; i < 16; ++i) {
            float4 v = ((const float4*)src)[i];
            dst[i*4+0] = (__bf16)v.x; dst[i*4+1] = (__bf16)v.y;
            dst[i*4+2] = (__bf16)v.z; dst[i*4+3] = (__bf16)v.w;
          }
        } else {
          #pragma unroll
          for (int i = 0; i < 64; ++i) dst[i] = (__bf16)0.f;
        }
      }
      { // stage B^T: Bs[ci][co] = wstd[2-kh, 2-kw, ci, co]
        const float* src = wstd + ((size_t)(((2 - kh) * 3 + (2 - kw)) * CH + r)) * CH + half * 64;
        __bf16* dst = &Bs[r][half * 64];
        #pragma unroll
        for (int i = 0; i < 16; ++i) {
          float4 v = ((const float4*)src)[i];
          dst[i*4+0] = (__bf16)v.x; dst[i*4+1] = (__bf16)v.y;
          dst[i*4+2] = (__bf16)v.z; dst[i*4+3] = (__bf16)v.w;
        }
      }
      __syncthreads();

      #pragma unroll
      for (int s = 0; s < 4; ++s) {
        v8bf a0 = *(const v8bf*)&As[arow][s * 32 + akb];
        v8bf a1 = *(const v8bf*)&As[arow][s * 32 + 16 + akb];
        v16bf a = __builtin_shufflevector(a0, a1, 0,1,2,3,4,5,6,7,8,9,10,11,12,13,14,15);
        v16bf bb[8];
        #pragma unroll
        for (int j = 0; j < 8; ++j) {
          v8bf b0 = *(const v8bf*)&Bs[j * 16 + nn][s * 32 + bkb];
          v8bf b1 = *(const v8bf*)&Bs[j * 16 + nn][s * 32 + bkb + 16];
          bb[j] = __builtin_shufflevector(b0, b1, 0,1,2,3,4,5,6,7,8,9,10,11,12,13,14,15);
        }
        #pragma unroll
        for (int j = 0; j < 8; ++j)
          acc[j] = __builtin_amdgcn_wmma_f32_16x16x32_bf16(false, a, false, bb[j],
                                                           (short)0, acc[j], false, false);
        mma_pipeline_hint();
      }
      __syncthreads();
    }
  }

  // ---- writeback via LDS: scatter accumulators, then coalesced b128 stores
  typedef float (*CT)[CPAD];
  CT Cf = (CT)smem;                       // 128*132*4 = 67584B <= 69632B
  const int rloc = wave * 16 + ((lane < 16) ? 0 : 8);
  const int cbase = lane & 15;
  #pragma unroll
  for (int j = 0; j < 8; ++j)
    #pragma unroll
    for (int v = 0; v < 8; ++v)
      Cf[rloc + v][j * 16 + cbase] = acc[j][v];
  __syncthreads();
  {
    const float* srcl = &Cf[r][half * 64];
    float* dstg = gin + (size_t)m * CH + half * 64;
    #pragma unroll
    for (int i = 0; i < 16; ++i)
      ((float4*)dstg)[i] = ((const float4*)srcl)[i];
  }
}

// ---------------------------------------------------------------------------
// wgrad raw: wraw[kh,kw,ci,co] += sum_p inputs[p shifted, ci] * go[p, co]
// Split-K over pixels; 128 pixels staged per round (32 WMMAs / barrier pair);
// f32 atomics accumulate into the zeroed output region.
// ---------------------------------------------------------------------------
__global__ __launch_bounds__(256) void wgrad_kernel(
    const float* __restrict__ inp, const float* __restrict__ go,
    float* __restrict__ wraw)
{
  __shared__ __bf16 At[128][APAD];   // [ci][pixel]
  __shared__ __bf16 Bt[128][APAD];   // [co][pixel]

  const int t    = threadIdx.x;
  const int wave = t >> 5, lane = t & 31;
  const int khw   = blockIdx.x / SPLIT;
  const int chunk = blockIdx.x - khw * SPLIT;
  const int kh = khw / 3, kw = khw - kh * 3;
  const int p0 = chunk * PIX_PER_BLOCK;

  v8f acc[8];
  #pragma unroll
  for (int j = 0; j < 8; ++j)
    #pragma unroll
    for (int v = 0; v < 8; ++v) acc[j][v] = 0.f;

  const int lp = t >> 1;          // local pixel 0..127
  const int c0 = (t & 1) * 64;    // channel half

  const int arow = wave * 16 + (lane & 15);
  const int akb  = (lane < 16) ? 0 : 8;
  const int bkb  = (lane < 16) ? 0 : 8;
  const int nn   = lane & 15;

  for (int it = 0; it < PIX_PER_BLOCK / 128; ++it) {
    const int p  = p0 + it * 128 + lp;
    const int b  = p / IMG;
    const int yx = p - b * IMG;
    const int y  = yx / HW;
    const int x  = yx - y * HW;
    const int sy = y + kh - 1, sx = x + kw - 1;
    const bool ok = (sy >= 0) && (sy < HW) && (sx >= 0) && (sx < HW);

    if (ok) { // inputs (shifted) -> At[ci][pixel] (transposed scatter)
      const float* src = inp + ((size_t)((b * HW + sy) * HW + sx)) * CH + c0;
      #pragma unroll
      for (int i = 0; i < 16; ++i) {
        float4 v = ((const float4*)src)[i];
        At[c0 + i*4 + 0][lp] = (__bf16)v.x;
        At[c0 + i*4 + 1][lp] = (__bf16)v.y;
        At[c0 + i*4 + 2][lp] = (__bf16)v.z;
        At[c0 + i*4 + 3][lp] = (__bf16)v.w;
      }
    } else {
      #pragma unroll
      for (int i = 0; i < 64; ++i) At[c0 + i][lp] = (__bf16)0.f;
    }
    { // grad_out -> Bt[co][pixel]
      const float* src = go + (size_t)p * CH + c0;
      #pragma unroll
      for (int i = 0; i < 16; ++i) {
        float4 v = ((const float4*)src)[i];
        Bt[c0 + i*4 + 0][lp] = (__bf16)v.x;
        Bt[c0 + i*4 + 1][lp] = (__bf16)v.y;
        Bt[c0 + i*4 + 2][lp] = (__bf16)v.z;
        Bt[c0 + i*4 + 3][lp] = (__bf16)v.w;
      }
    }
    __syncthreads();

    #pragma unroll
    for (int s = 0; s < 4; ++s) {
      v8bf a0 = *(const v8bf*)&At[arow][s * 32 + akb];
      v8bf a1 = *(const v8bf*)&At[arow][s * 32 + 16 + akb];
      v16bf a = __builtin_shufflevector(a0, a1, 0,1,2,3,4,5,6,7,8,9,10,11,12,13,14,15);
      v16bf bb[8];
      #pragma unroll
      for (int j = 0; j < 8; ++j) {
        v8bf b0 = *(const v8bf*)&Bt[j * 16 + nn][s * 32 + bkb];
        v8bf b1 = *(const v8bf*)&Bt[j * 16 + nn][s * 32 + bkb + 16];
        bb[j] = __builtin_shufflevector(b0, b1, 0,1,2,3,4,5,6,7,8,9,10,11,12,13,14,15);
      }
      #pragma unroll
      for (int j = 0; j < 8; ++j)
        acc[j] = __builtin_amdgcn_wmma_f32_16x16x32_bf16(false, a, false, bb[j],
                                                         (short)0, acc[j], false, false);
      mma_pipeline_hint();
    }
    __syncthreads();
  }

  const int cib = wave * 16 + ((lane < 16) ? 0 : 8);   // ci base (M)
  const int cob = lane & 15;                            // co base (N)
  #pragma unroll
  for (int j = 0; j < 8; ++j)
    #pragma unroll
    for (int v = 0; v < 8; ++v)
      unsafeAtomicAdd(&wraw[((size_t)(khw * CH + cib + v)) * CH + j * 16 + cob],
                      acc[j][v]);
}

// ---------------------------------------------------------------------------
// bias: sum over all pixels of grad_output, per channel
// ---------------------------------------------------------------------------
__global__ __launch_bounds__(256) void bias_kernel(const float* __restrict__ go,
                                                   float* __restrict__ bias)
{
  const int c = threadIdx.x & 127;
  const int g = threadIdx.x >> 7;            // 0 or 1
  const int pbase = blockIdx.x * 256;
  float s = 0.f;
  for (int p = pbase + g; p < pbase + 256; p += 2)
    s += go[(size_t)p * CH + c];
  unsafeAtomicAdd(&bias[c], s);
}

// ---------------------------------------------------------------------------
// weight-standardization backward chain; in-place on wraw -> grad_wt
// ---------------------------------------------------------------------------
__device__ __forceinline__ float block_reduce128(float v, float* red)
{
  const int t = threadIdx.x;
  __syncthreads();
  red[t] = v;
  __syncthreads();
  #pragma unroll
  for (int off = 64; off > 0; off >>= 1) {
    if (t < off) red[t] += red[t + off];
    __syncthreads();
  }
  return red[0];
}

__global__ __launch_bounds__(128) void wschain_kernel(
    float* __restrict__ wt,                   // in: raw grad, out: final grad
    const float* __restrict__ kern,           // [3,3,Cin,Cout]
    const float* __restrict__ gain,
    const float* __restrict__ bitmap,
    float* __restrict__ out_gain)
{
  __shared__ float red[128];
  const int co = blockIdx.x;
  const int t  = threadIdx.x;
  const float Nf = 9.f * 128.f;               // 1152

  // pass 1: mean of kernels[:, :, :, co]
  float s = 0.f;
  float w[9], gr[9];
  #pragma unroll
  for (int i = 0; i < 9; ++i) {
    w[i] = kern[(size_t)(t + i * 128) * CH + co];
    s += w[i];
  }
  const float mu = block_reduce128(s, red) / Nf;

  // pass 2: variance sum and dgivar = sum(graw * xmu)
  float s2 = 0.f, sg = 0.f;
  #pragma unroll
  for (int i = 0; i < 9; ++i) {
    gr[i] = wt[(size_t)(t + i * 128) * CH + co];
    const float xm = w[i] - mu;
    w[i] = xm;                                // keep xmu
    s2 += xm * xm;
    sg += gr[i] * xm;
  }
  const float S2 = block_reduce128(s2, red);
  const float dgivar = block_reduce128(sg, red);

  const float var      = S2 / Nf;
  const float mvar     = Nf * var;
  const float maxvar   = fmaxf(mvar, 1e-4f);
  const float sqrtvar  = sqrtf(maxvar);
  const float ivar     = 1.f / sqrtvar;
  const float gv       = gain[co];
  const float givar    = gv * ivar;
  const float divar    = dgivar * gv;
  const float dgain    = dgivar * ivar;
  const float dsqrtvar = (-1.f / (sqrtvar * sqrtvar)) * divar;
  const float dmaxvar  = (0.5f / sqrtvar) * dsqrtvar;
  const float dmvar    = bitmap[co] * dmaxvar;
  const float dvar     = Nf * dmvar;
  const float dsq      = dvar / Nf;

  // pass 3: dx1 and dmu
  float dx1[9], sd = 0.f;
  #pragma unroll
  for (int i = 0; i < 9; ++i) {
    dx1[i] = gr[i] * givar + 2.f * w[i] * dsq;
    sd += dx1[i];
  }
  const float dmu = -block_reduce128(sd, red);
  const float dx2 = dmu / Nf;

  // pass 4: final write (each thread overwrites only what it read)
  #pragma unroll
  for (int i = 0; i < 9; ++i)
    wt[(size_t)(t + i * 128) * CH + co] = dx1[i] + dx2;

  if (t == 0) out_gain[co] = dgain;
}

// ---------------------------------------------------------------------------
extern "C" void kernel_launch(void* const* d_in, const int* in_sizes, int n_in,
                              void* d_out, int out_size, void* d_ws, size_t ws_size,
                              hipStream_t stream)
{
  const float* go   = (const float*)d_in[0];   // grad_output [64,56,56,128]
  const float* inp  = (const float*)d_in[1];   // inputs      [64,56,56,128]
  const float* kern = (const float*)d_in[2];   // kernels     [3,3,128,128]
  const float* wstd = (const float*)d_in[3];   // std_kernels [3,3,128,128]
  const float* gain = (const float*)d_in[4];   // [128]
  const float* bmap = (const float*)d_in[5];   // [128]

  float* out_gin  = (float*)d_out;                       // 200704*128
  float* out_wt   = out_gin + (size_t)MTOT * CH;         // 9*128*128
  float* out_bias = out_wt + (size_t)9 * CH * CH;        // 128
  float* out_gain = out_bias + CH;                       // 128

  // zero the atomically-accumulated regions (grad_wt raw + bias)
  hipMemsetAsync(out_wt, 0, ((size_t)9 * CH * CH + CH) * sizeof(float), stream);

  dgrad_kernel<<<MTOT / 128, 256, 0, stream>>>(go, wstd, out_gin);
  wgrad_kernel<<<9 * SPLIT, 256, 0, stream>>>(inp, go, out_wt);
  bias_kernel<<<MTOT / 256, 256, 0, stream>>>(go, out_bias);
  wschain_kernel<<<CH, 128, 0, stream>>>(out_wt, kern, gain, bmap, out_gain);
}
